// DConvMotionCompensation_50534585205118
// MI455X (gfx1250) — matrix operand order repaired and checked by reading, not compiled, not run on hardware
//
#include <hip/hip_runtime.h>

// ---------------------------------------------------------------------------
// Problem constants (match reference)
// ---------------------------------------------------------------------------
#define BB   8
#define CC   64
#define HH   128
#define WW   128
#define HW   16384          // 128*128
#define KTAP 9

// padded NHWC activation geometry (1-pixel zero halo)
#define WP   130
#define RS   (WP * CC)            // row stride in elements  (8320)
#define IS   (WP * WP * CC)       // image stride in elements (1081600)

typedef __bf16 v16bf __attribute__((ext_vector_type(16)));
typedef float  v8f   __attribute__((ext_vector_type(8)));
typedef unsigned int u32x4 __attribute__((ext_vector_type(4)));

union BFrag {
    v16bf v;
    unsigned short u[16];
    u32x4 q[2];
};

__device__ __forceinline__ unsigned short f2bf(float f) {
    unsigned int u = __float_as_uint(f);
    unsigned int r = u + 0x7FFFu + ((u >> 16) & 1u);   // round to nearest even
    return (unsigned short)(r >> 16);
}
__device__ __forceinline__ float bf2f(unsigned short s) {
    return __uint_as_float(((unsigned int)s) << 16);
}

// ---------------------------------------------------------------------------
// LDS weight staging: async copy (CDNA5 GLOBAL_LOAD_ASYNC_TO_LDS_B128) when
// the toolchain exposes it, VGPR-roundtrip copy otherwise. 4608 x 16B = 72KB.
// Builtin signature (from hipcc diagnostic): param0 = AS(1) int4*, non-const.
// ---------------------------------------------------------------------------
__device__ __forceinline__ void stage_weights(unsigned short* lds_w,
                                              const unsigned short* pw,
                                              int tid) {
#if defined(__gfx1250__) && defined(__has_builtin) && \
    __has_builtin(__builtin_amdgcn_global_load_async_to_lds_b128)
    typedef int vsi4 __attribute__((vector_size(16)));
    typedef __attribute__((address_space(1))) vsi4 gvsi4;
    typedef __attribute__((address_space(3))) vsi4 lvsi4;
    gvsi4* g = (gvsi4*)(unsigned long long)(const void*)pw;
    lvsi4* l = (lvsi4*)(unsigned)(unsigned long long)(void*)lds_w;
    for (int i = tid; i < 4608; i += 256)
        __builtin_amdgcn_global_load_async_to_lds_b128(g + i, l + i, 0, 0);
#if __has_builtin(__builtin_amdgcn_s_wait_asynccnt)
    __builtin_amdgcn_s_wait_asynccnt(0);
#else
    asm volatile("s_wait_asynccnt 0x0" ::: "memory");
#endif
#else
    u32x4* l4 = (u32x4*)lds_w;
    const u32x4* g4 = (const u32x4*)pw;
    for (int i = tid; i < 4608; i += 256) l4[i] = g4[i];
#endif
    __syncthreads();
}

// ---------------------------------------------------------------------------
// Pack f32 weights [64,64,3,3] into bf16 WMMA A-fragments.
// Packed layout: frag f = (t*2+kb)*4 + mt  (step-major!) so each K-step's
// four M-tile fragments are one contiguous 4KB run -> ds_load immediates.
// Element offset = (f*32+lane)*16 + e.
// A 16x32 bf16 layout: m = lane%16 ; k = 16*(lane/16) + e
// ---------------------------------------------------------------------------
__global__ void pack_w_kernel(const float* __restrict__ w,
                              unsigned short* __restrict__ dst) {
    int idx = blockIdx.x * 256 + threadIdx.x;          // < 72*512 = 36864
    if (idx >= 72 * 512) return;
    int e    = idx & 15;
    int lane = (idx >> 4) & 31;
    int f    = idx >> 9;           // 0..71, step-major
    int step = f >> 2;             // 0..17
    int mt   = f & 3;
    int t    = step >> 1;
    int kb   = step & 1;
    int m    = lane & 15;
    int lh   = lane >> 4;
    int k    = 16 * lh + e;
    int co   = mt * 16 + m;
    int ci   = kb * 32 + k;
    dst[idx] = f2bf(w[(co * CC + ci) * KTAP + t]);
}

// ---------------------------------------------------------------------------
// NCHW f32 -> padded NHWC bf16 (interior only; halo pre-zeroed by memset)
// ---------------------------------------------------------------------------
__global__ void to_nhwc_bf16(const float* __restrict__ x,
                             unsigned short* __restrict__ out) {
    int idx = blockIdx.x * 256 + threadIdx.x;          // B*C*HW = 2^23
    int p   = idx & (HW - 1);
    int c   = (idx >> 14) & (CC - 1);
    int b   = idx >> 20;
    int yy  = p >> 7;
    int xx  = p & (WW - 1);
    out[(size_t)b * IS + ((yy + 1) * WP + (xx + 1)) * CC + c] = f2bf(x[idx]);
}

// ---------------------------------------------------------------------------
// 1x1 offset conv: of[B,2,HW] -> offs[B,18,HW]
// ---------------------------------------------------------------------------
__global__ void offset_conv(const float* __restrict__ of,
                            const float* __restrict__ w,
                            const float* __restrict__ bo,
                            float* __restrict__ out) {
    int idx = blockIdx.x * 256 + threadIdx.x;          // B*HW = 131072
    int p   = idx & (HW - 1);
    int b   = idx >> 14;
    float i0 = of[((size_t)(b * 2 + 0) << 14) + p];
    float i1 = of[((size_t)(b * 2 + 1) << 14) + p];
#pragma unroll
    for (int oc = 0; oc < 18; ++oc) {
        out[((size_t)(b * 18 + oc) << 14) + p] =
            fmaf(w[oc * 2 + 0], i0, fmaf(w[oc * 2 + 1], i1, bo[oc]));
    }
}

// ---------------------------------------------------------------------------
// 3x3 conv + bias + ReLU via implicit-GEMM bf16 WMMA on padded NHWC.
// Wave computes 64(Cout) x 32(pos): two B-fragments per K-step, each LDS
// A-fragment feeds two back-to-back WMMAs (ds:wmma = 1:1).
// Block = 8 waves = 256 positions = two image rows.
// ---------------------------------------------------------------------------
__global__ __launch_bounds__(256) void conv3x3_wmma(
        const unsigned short* __restrict__ act,    // padded NHWC bf16
        const unsigned short* __restrict__ pw,     // packed A-fragments (72KB)
        const float* __restrict__ bias,
        unsigned short* __restrict__ out)          // padded NHWC bf16 (relu'd)
{
    extern __shared__ unsigned short lds_w[];
    const int tid  = threadIdx.x;
    const int lane = tid & 31;
    const int wave = tid >> 5;
    const int b    = blockIdx.x >> 6;
    const int y    = ((blockIdx.x & 63) << 1) + (wave >> 2);   // wave's row
    const int x0   = (wave & 3) << 5;                          // wave's 32-pel span
    const int n    = lane & 15;
    const int lh   = lane >> 4;

    // lane's padded-center pointer for fragment 0; frag 1 = +16 pixels (+2KB)
    const unsigned short* base =
        act + (size_t)b * IS + ((y + 1) * WP + (x0 + n + 1)) * CC + 16 * lh;

    // warm the three tap rows for both fragments while we fill LDS
    __builtin_prefetch(base - RS, 0, 0);
    __builtin_prefetch(base,      0, 0);
    __builtin_prefetch(base + RS, 0, 0);
    __builtin_prefetch(base - RS + 16 * CC, 0, 0);
    __builtin_prefetch(base      + 16 * CC, 0, 0);
    __builtin_prefetch(base + RS + 16 * CC, 0, 0);

    stage_weights(lds_w, pw, tid);

    v8f acc[4][2] = {};

#pragma unroll
    for (int t = 0; t < KTAP; ++t) {
        const int dy = t / 3 - 1;
        const int dx = t % 3 - 1;
        const unsigned short* src = base + (dy * WP + dx) * CC;
#pragma unroll
        for (int kb = 0; kb < 2; ++kb) {
            BFrag bf0, bf1;
            const u32x4* s4 = (const u32x4*)(src + kb * 32);
            bf0.q[0] = s4[0];
            bf0.q[1] = s4[1];
            const u32x4* s4b = (const u32x4*)(src + kb * 32 + 16 * CC);
            bf1.q[0] = s4b[0];
            bf1.q[1] = s4b[1];
            const int step = t * 2 + kb;
            const unsigned short* lbase = lds_w + ((step * 4 * 32 + lane) << 4);
#pragma unroll
            for (int mt = 0; mt < 4; ++mt) {
                BFrag af;
                const u32x4* l4 = (const u32x4*)(lbase + (mt << 9));
                af.q[0] = l4[0];
                af.q[1] = l4[1];
                acc[mt][0] = __builtin_amdgcn_wmma_f32_16x16x32_bf16(
                    false, af.v, false, bf0.v, (short)0, acc[mt][0], false, false);
                acc[mt][1] = __builtin_amdgcn_wmma_f32_16x16x32_bf16(
                    false, af.v, false, bf1.v, (short)0, acc[mt][1], false, false);
            }
        }
    }

    // bias + relu + store: per M-tile/fragment, 8 channels -> one b128 store
    unsigned short* outp =
        out + (size_t)b * IS + ((y + 1) * WP + (x0 + n + 1)) * CC;
#pragma unroll
    for (int f = 0; f < 2; ++f) {
#pragma unroll
        for (int mt = 0; mt < 4; ++mt) {
            const int co0 = mt * 16 + 8 * lh;
            BFrag o;
#pragma unroll
            for (int r = 0; r < 8; ++r) {
                float v = acc[mt][f][r] + bias[co0 + r];
                o.u[r] = f2bf(v > 0.f ? v : 0.f);
            }
            *(u32x4*)(outp + f * 16 * CC + co0) = o.q[0];
        }
    }
}

// ---------------------------------------------------------------------------
// Deformable 3x3 conv: bilinear gather (f32 blend of bf16 padded-NHWC
// activations) -> bf16 B-fragments -> same 64x32-per-wave WMMA chain;
// output f32 NCHW (+bias). Corner validity checks kept (data-dep offsets).
// ---------------------------------------------------------------------------
__global__ __launch_bounds__(256) void deform_wmma(
        const unsigned short* __restrict__ act,   // h, padded NHWC bf16
        const float* __restrict__ off,            // [B,18,HW] f32
        const unsigned short* __restrict__ pw,    // packed w_d fragments
        const float* __restrict__ bias,
        float* __restrict__ out)                  // [B,64,HW] f32
{
    extern __shared__ unsigned short lds_w[];
    const int tid = threadIdx.x;
    stage_weights(lds_w, pw, tid);

    const int lane = tid & 31;
    const int wave = tid >> 5;
    const int b    = blockIdx.x >> 6;
    const int y    = ((blockIdx.x & 63) << 1) + (wave >> 2);
    const int x0   = (wave & 3) << 5;
    const int n    = lane & 15;
    const int lh   = lane >> 4;

    v8f acc[4][2] = {};

    const unsigned short* actb = act + (size_t)b * IS;
    const float* offb = off + (((size_t)b * 18) << 14);

    for (int t = 0; t < KTAP; ++t) {
        // per-fragment sampling geometry (position differs by 16 pixels)
        int   iy[2], ix[2];
        float wy[2][2], wx[2][2];
#pragma unroll
        for (int f = 0; f < 2; ++f) {
            const int xf = x0 + f * 16 + n;
            const int pf = (y << 7) + xf;
            const float py = (float)(y + t / 3 - 1) + offb[((size_t)(2 * t + 0) << 14) + pf];
            const float px = (float)(xf + t % 3 - 1) + offb[((size_t)(2 * t + 1) << 14) + pf];
            const float fy0 = floorf(py);
            const float fx0 = floorf(px);
            iy[f] = (int)fy0;
            ix[f] = (int)fx0;
            const float fy = py - fy0;
            const float fx = px - fx0;
            wy[f][0] = 1.f - fy; wy[f][1] = fy;
            wx[f][0] = 1.f - fx; wx[f][1] = fx;
        }

#pragma unroll
        for (int kb = 0; kb < 2; ++kb) {
            const int c0 = kb * 32 + 16 * lh;
            BFrag bfr[2];
#pragma unroll
            for (int f = 0; f < 2; ++f) {
                float a16[16];
#pragma unroll
                for (int i = 0; i < 16; ++i) a16[i] = 0.f;
#pragma unroll
                for (int cy = 0; cy < 2; ++cy) {
#pragma unroll
                    for (int cx = 0; cx < 2; ++cx) {
                        const int yc = iy[f] + cy;
                        const int xc = ix[f] + cx;
                        if (yc >= 0 && yc < HH && xc >= 0 && xc < WW) {
                            const float wgt = wy[f][cy] * wx[f][cx];
                            const u32x4* s4 = (const u32x4*)
                                (actb + ((yc + 1) * WP + (xc + 1)) * CC + c0);
                            BFrag v;
                            v.q[0] = s4[0];
                            v.q[1] = s4[1];
#pragma unroll
                            for (int i = 0; i < 16; ++i)
                                a16[i] = fmaf(wgt, bf2f(v.u[i]), a16[i]);
                        }
                    }
                }
#pragma unroll
                for (int i = 0; i < 16; ++i) bfr[f].u[i] = f2bf(a16[i]);
            }
            const int step = t * 2 + kb;
            const unsigned short* lbase = lds_w + ((step * 4 * 32 + lane) << 4);
#pragma unroll
            for (int mt = 0; mt < 4; ++mt) {
                BFrag af;
                const u32x4* l4 = (const u32x4*)(lbase + (mt << 9));
                af.q[0] = l4[0];
                af.q[1] = l4[1];
                acc[mt][0] = __builtin_amdgcn_wmma_f32_16x16x32_bf16(
                    false, af.v, false, bfr[0].v, (short)0, acc[mt][0], false, false);
                acc[mt][1] = __builtin_amdgcn_wmma_f32_16x16x32_bf16(
                    false, af.v, false, bfr[1].v, (short)0, acc[mt][1], false, false);
            }
        }
    }

    float* outb = out + ((size_t)(b << 6) << 14);
#pragma unroll
    for (int f = 0; f < 2; ++f) {
        const int pf = (y << 7) + x0 + f * 16 + n;
#pragma unroll
        for (int mt = 0; mt < 4; ++mt) {
            const int co0 = mt * 16 + 8 * lh;
#pragma unroll
            for (int r = 0; r < 8; ++r) {
                outb[((size_t)(co0 + r) << 14) + pf] = acc[mt][f][r] + bias[co0 + r];
            }
        }
    }
}

// ---------------------------------------------------------------------------
// Launcher. setup_inputs order:
// 0:x 1:of 2:w1 3:b1 4:w2 5:b2 6:w3 7:b3 8:w_off 9:b_off 10:w_d 11:b_d
// ---------------------------------------------------------------------------
extern "C" void kernel_launch(void* const* d_in, const int* in_sizes, int n_in,
                              void* d_out, int out_size, void* d_ws, size_t ws_size,
                              hipStream_t stream) {
    (void)in_sizes; (void)n_in; (void)out_size; (void)ws_size;

    const float* x     = (const float*)d_in[0];
    const float* of    = (const float*)d_in[1];
    const float* w1    = (const float*)d_in[2];
    const float* b1    = (const float*)d_in[3];
    const float* w2    = (const float*)d_in[4];
    const float* b2    = (const float*)d_in[5];
    const float* w3    = (const float*)d_in[6];
    const float* b3    = (const float*)d_in[7];
    const float* w_off = (const float*)d_in[8];
    const float* b_off = (const float*)d_in[9];
    const float* w_d   = (const float*)d_in[10];
    const float* b_d   = (const float*)d_in[11];
    float* out = (float*)d_out;

    // workspace layout (padded activations)
    char* ws = (char*)d_ws;
    const size_t ACT_BYTES = (size_t)BB * IS * 2;               // 17,305,600 B
    unsigned short* actA = (unsigned short*)ws;
    unsigned short* actB = (unsigned short*)(ws + ACT_BYTES);
    unsigned short* pw   = (unsigned short*)(ws + 2 * ACT_BYTES);
    const size_t PW_ELEMS = 72 * 512;                           // 36864 per set
    float* offs = (float*)(ws + 2 * ACT_BYTES + 4 * PW_ELEMS * 2);

    const size_t SHM = 4608 * 16;                               // 72 KB packed A

    // zero both padded activation buffers (establishes the zero halo)
    (void)hipMemsetAsync(ws, 0, 2 * ACT_BYTES, stream);

    pack_w_kernel<<<144, 256, 0, stream>>>(w1,  pw + 0 * PW_ELEMS);
    pack_w_kernel<<<144, 256, 0, stream>>>(w2,  pw + 1 * PW_ELEMS);
    pack_w_kernel<<<144, 256, 0, stream>>>(w3,  pw + 2 * PW_ELEMS);
    pack_w_kernel<<<144, 256, 0, stream>>>(w_d, pw + 3 * PW_ELEMS);

    to_nhwc_bf16<<<(BB * CC * HW) / 256, 256, 0, stream>>>(x, actA);
    offset_conv<<<(BB * HW) / 256, 256, 0, stream>>>(of, w_off, b_off, offs);

    conv3x3_wmma<<<BB * 64, 256, SHM, stream>>>(actA, pw + 0 * PW_ELEMS, b1, actB);
    conv3x3_wmma<<<BB * 64, 256, SHM, stream>>>(actB, pw + 1 * PW_ELEMS, b2, actA);
    conv3x3_wmma<<<BB * 64, 256, SHM, stream>>>(actA, pw + 2 * PW_ELEMS, b3, actB);

    deform_wmma<<<BB * 64, 256, SHM, stream>>>(actB, offs, pw + 3 * PW_ELEMS, b_d, out);
}
